// NeurologicalAttention_28887950033390
// MI455X (gfx1250) — matrix operand order repaired and checked by reading, not compiled, not run on hardware
//
#include <hip/hip_runtime.h>
#include <math.h>

// ---------------- CDNA5 WMMA types ----------------
typedef __attribute__((ext_vector_type(16))) __bf16 v16bf;
typedef __attribute__((ext_vector_type(8)))  float  v8f;

#define SEQ      2048
#define MODELDIM 1024
#define NHEADS   16
#define HDIM     64
#define TOPK     204            // max(1, int(2048*0.1))
#define MINV     (-3.402823466e+38f)

// ---------------- helpers ----------------
__device__ __forceinline__ unsigned short f2bf(float f) {
    union { float f; unsigned u; } v; v.f = f;
    unsigned r = v.u + 0x7FFFu + ((v.u >> 16) & 1u);   // RNE
    return (unsigned short)(r >> 16);
}
__device__ __forceinline__ unsigned pack2(float a, float b) {
    return (unsigned)f2bf(a) | ((unsigned)f2bf(b) << 16);
}
// monotone float->uint key (total order)
__device__ __forceinline__ unsigned f2key(float f) {
    union { float f; unsigned u; } v; v.f = f;
    return (v.u & 0x80000000u) ? ~v.u : (v.u | 0x80000000u);
}
__device__ __forceinline__ float key2f(unsigned k) {
    union { float f; unsigned u; } v;
    v.u = (k & 0x80000000u) ? (k & 0x7FFFFFFFu) : ~k;
    return v.f;
}

union FragBF { v16bf v; unsigned u[8]; };

// ---- CDNA5 async global->LDS DMA (ASYNCcnt-tracked, ISA 10.7/15.18) ----
__device__ __forceinline__ void async_b128(unsigned lds_off, const void* g) {
    asm volatile("global_load_async_to_lds_b128 %0, %1, off"
                 :: "v"(lds_off), "v"((unsigned long long)g) : "memory");
}
__device__ __forceinline__ void wait_async0() {
    asm volatile("s_wait_asynccnt 0" ::: "memory");
}
__device__ __forceinline__ void wait_async4() {
    asm volatile("s_wait_asynccnt 4" ::: "memory");
}
__device__ __forceinline__ unsigned lds_off_of(const void* p) {
    return (unsigned)(unsigned long long)p;   // low 32b of flat LDS addr = ds offset
}

// =====================================================================
// One-shot f32 -> packed-bf16 conversion (pairs)
// =====================================================================
__global__ __launch_bounds__(256) void f32_to_bf16_pk(
    const float* __restrict__ src, unsigned* __restrict__ dst, int n2)
{
    int i = blockIdx.x * 256 + threadIdx.x;
    if (i < n2) {
        float2 v = ((const float2*)src)[i];
        dst[i] = pack2(v.x, v.y);
    }
}

// =====================================================================
// GEMM: C[M][N] = A[M][K] @ Bw[N][K]^T (+bias), bf16 operands, WMMA.
// 128 thr (4 waves), tile 64x64, Ktile 32, double-buffered async-to-LDS.
// =====================================================================
__global__ __launch_bounds__(128) void gemm_bf16_async(
    const unsigned short* __restrict__ A, const unsigned short* __restrict__ Bw,
    const float* __restrict__ bias, float* __restrict__ C,
    int M, int N, int K)
{
    __shared__ unsigned ldsA[2][1024];   // [buf][row][kpair]
    __shared__ unsigned ldsB[2][1024];   // [buf][n][kpair]
    const int tid  = threadIdx.x;
    const int lane = tid & 31, wv = tid >> 5;
    const int ln   = lane & 15, hi = lane >> 4;
    const int m0 = blockIdx.y * 64, n0 = blockIdx.x * 64;
    const unsigned baseA = lds_off_of(&ldsA[0][0]);
    const unsigned baseB = lds_off_of(&ldsB[0][0]);
    v8f acc[4] = {};

    auto issue = [&](int buf, int k0) {
        // tile = 64 rows x 32 bf16 (64B) = 4 x b128 chunks per row
        #pragma unroll
        for (int rep = 0; rep < 2; rep++) {
            int q = tid + rep * 128;               // 256 chunks per operand
            int row = q >> 2, c4 = q & 3;
            async_b128(baseA + (unsigned)(buf * 4096 + q * 16),
                       &A [(size_t)(m0 + row) * K + k0 + c4 * 8]);
            async_b128(baseB + (unsigned)(buf * 4096 + q * 16),
                       &Bw[(size_t)(n0 + row) * K + k0 + c4 * 8]);
        }
    };

    const int nIter = K >> 5;
    issue(0, 0);
    for (int it = 0; it < nIter; it++) {
        const int buf = it & 1;
        if (it + 1 < nIter) { issue(buf ^ 1, (it + 1) << 5); wait_async4(); }
        else                { wait_async0(); }
        __syncthreads();
        const unsigned* lA = ldsA[buf];
        const unsigned* lB = ldsB[buf];
        FragBF a;
        const int ra = (wv * 16 + ln) * 16;
        #pragma unroll
        for (int i = 0; i < 4; i++) {
            a.u[i]     = lA[ra + hi * 8 + i];          // K 0..7 | 16..23
            a.u[4 + i] = lA[ra + hi * 8 + 4 + i];      // K 8..15 | 24..31
        }
        #pragma unroll
        for (int j = 0; j < 4; j++) {
            FragBF b;
            const int rb = (j * 16 + ln) * 16 + hi * 8;
            #pragma unroll
            for (int i = 0; i < 8; i++) b.u[i] = lB[rb + i];
            acc[j] = __builtin_amdgcn_wmma_f32_16x16x32_bf16(
                false, a.v, false, b.v, (short)0, acc[j], false, false);
        }
        __syncthreads();
    }
    #pragma unroll
    for (int j = 0; j < 4; j++) {
        int col = n0 + j * 16 + ln;
        float bv = bias ? bias[col] : 0.f;
        #pragma unroll
        for (int r = 0; r < 8; r++) {
            int m = m0 + wv * 16 + hi * 8 + r;
            C[(size_t)m * N + col] = acc[j][r] + bv;
        }
    }
}

// =====================================================================
// Fused dendritic conv + oscillatory gain + RoPE, f32->bf16 head-major.
// One thread per (t, channel-pair). scale=1/8 folded into Q (exact).
// =====================================================================
__global__ __launch_bounds__(256) void dendro_rope_kernel(
    const float* __restrict__ qkv,
    const float* __restrict__ kq, const float* __restrict__ kk,
    const float* __restrict__ kv,
    const float* __restrict__ gf, const float* __restrict__ tf,
    const float* __restrict__ ph,
    unsigned* __restrict__ qO, unsigned* __restrict__ kO,
    unsigned* __restrict__ vO)
{
    int gid = blockIdx.x * 256 + threadIdx.x;      // [0, 2048*512)
    int t = gid >> 9;
    int c = (gid & 511) << 1;

    float q0 = 0.f, q1 = 0.f, kk0 = 0.f, kk1 = 0.f, v0 = 0.f, v1 = 0.f;
    #pragma unroll
    for (int i = 0; i < 4; i++) {
        int s = t - 3 + i;                         // out[t] = sum k[i]*x[t-3+i]
        if (s >= 0) {
            const float* r = qkv + (size_t)s * 3072;
            float wq = kq[i], wk = kk[i], wvv = kv[i];
            q0  += wq  * r[c];          q1  += wq  * r[c + 1];
            kk0 += wk  * r[1024 + c];   kk1 += wk  * r[1024 + c + 1];
            v0  += wvv * r[2048 + c];   v1  += wvv * r[2048 + c + 1];
        }
    }
    const float TP = 6.28318530717958647692f;
    float tpos = (float)t;
    float th = 0.2f * __cosf(TP * tf[0] * tpos * 0.01f);
    float g0 = 0.5f + 0.3f * __cosf(TP * gf[c]     * tpos * 0.01f + ph[c])     + th;
    float g1 = 0.5f + 0.3f * __cosf(TP * gf[c + 1] * tpos * 0.01f + ph[c + 1]) + th;
    q0 *= g0; q1 *= g1; kk0 *= g0; kk1 *= g1;

    int h = c >> 6, d = c & 63;
    float invf = __expf(-(float)d * (9.21034037197618f / 64.f)); // 10000^(-d/64)
    float ang = tpos * invf;
    float cs = __cosf(ang), sn = __sinf(ang);
    float qe = q0 * cs - q1 * sn,  qo = q0 * sn + q1 * cs;
    float ke = kk0 * cs - kk1 * sn, ko = kk0 * sn + kk1 * cs;
    qe *= 0.125f; qo *= 0.125f;                    // fold softmax scale

    int o = (h * SEQ + t) * 32 + (d >> 1);         // [h][t][d] bf16, packed
    qO[o] = pack2(qe, qo);
    kO[o] = pack2(ke, ko);
    vO[o] = pack2(v0, v1);
}

// =====================================================================
// scores[m][n] = Qh[m][:] . Kh[n][:] (bf16 WMMA, K=64), causal mask.
// Full K-panels loaded in one async burst; masked tiles skip compute.
// =====================================================================
__global__ __launch_bounds__(128) void scores_kernel(
    const unsigned short* __restrict__ Q, const unsigned short* __restrict__ Kh,
    float* __restrict__ S)
{
    const int tid = threadIdx.x;
    const int m0 = blockIdx.y * 64, n0 = blockIdx.x * 64;
    if (n0 > m0 + 63) {                             // fully above diagonal
        for (int u = tid; u < 4096; u += 128) {
            int r = u >> 6, cc = u & 63;
            S[(size_t)(m0 + r) * SEQ + n0 + cc] = MINV;
        }
        return;
    }
    __shared__ unsigned ldsA[2048];   // [row][32 dwords] (full K=64)
    __shared__ unsigned ldsB[2048];
    const int lane = tid & 31, wv = tid >> 5;
    const int ln = lane & 15, hi = lane >> 4;
    const unsigned baseA = lds_off_of(&ldsA[0]);
    const unsigned baseB = lds_off_of(&ldsB[0]);

    #pragma unroll
    for (int rep = 0; rep < 4; rep++) {            // 512 chunks per operand
        int q = tid + rep * 128;
        int row = q >> 3, c8 = q & 7;              // 8 x b128 per 64-bf16 row
        async_b128(baseA + (unsigned)(q * 16), &Q [(size_t)(m0 + row) * 64 + c8 * 8]);
        async_b128(baseB + (unsigned)(q * 16), &Kh[(size_t)(n0 + row) * 64 + c8 * 8]);
    }
    wait_async0();
    __syncthreads();

    v8f acc[4] = {};
    #pragma unroll
    for (int k0 = 0; k0 < 64; k0 += 32) {
        FragBF a;
        const int ra = (wv * 16 + ln) * 32 + (k0 >> 1);
        #pragma unroll
        for (int i = 0; i < 4; i++) {
            a.u[i]     = ldsA[ra + hi * 8 + i];
            a.u[4 + i] = ldsA[ra + hi * 8 + 4 + i];
        }
        #pragma unroll
        for (int j = 0; j < 4; j++) {
            FragBF b;
            const int rb = (j * 16 + ln) * 32 + (k0 >> 1) + hi * 8;
            #pragma unroll
            for (int i = 0; i < 8; i++) b.u[i] = ldsB[rb + i];
            acc[j] = __builtin_amdgcn_wmma_f32_16x16x32_bf16(
                false, a.v, false, b.v, (short)0, acc[j], false, false);
        }
    }
    #pragma unroll
    for (int j = 0; j < 4; j++) {
        int col = n0 + j * 16 + ln;
        #pragma unroll
        for (int r = 0; r < 8; r++) {
            int m = m0 + wv * 16 + hi * 8 + r;
            S[(size_t)m * SEQ + col] = (col <= m) ? acc[j][r] : MINV;
        }
    }
}

// =====================================================================
// Per-row top-k(204) threshold + softmax -> bf16 probs.
// 8 elements/thread in registers; 32-step binary select over monotone
// uint keys; deterministic (int atomics + tree float reduction).
// =====================================================================
__global__ __launch_bounds__(256) void topk_softmax_kernel(
    const float* __restrict__ S, unsigned short* __restrict__ P)
{
    const int row = blockIdx.x, tid = threadIdx.x;
    float vals[8]; unsigned keys[8];
    #pragma unroll
    for (int i = 0; i < 8; i++) {
        float f = S[(size_t)row * SEQ + tid + (i << 8)];
        vals[i] = f; keys[i] = f2key(f);
    }
    __shared__ unsigned cnt, mk;
    __shared__ float sred[256];

    unsigned lo = 0u, hi = 0xFFFFFFFFu;
    while (lo < hi) {
        unsigned mid = lo + ((hi - lo) >> 1) + 1u;
        if (tid == 0) cnt = 0u;
        __syncthreads();
        unsigned c = 0u;
        #pragma unroll
        for (int i = 0; i < 8; i++) c += (keys[i] >= mid) ? 1u : 0u;
        if (c) atomicAdd(&cnt, c);
        __syncthreads();
        unsigned total = cnt;
        __syncthreads();
        if (total >= TOPK) lo = mid; else hi = mid - 1u;
    }
    const unsigned kth = lo;

    if (tid == 0) mk = 0u;
    __syncthreads();
    unsigned lm = 0u;
    #pragma unroll
    for (int i = 0; i < 8; i++) lm = (keys[i] > lm) ? keys[i] : lm;
    atomicMax(&mk, lm);
    __syncthreads();
    const float rowmax = key2f(mk);

    float ls = 0.f;
    #pragma unroll
    for (int i = 0; i < 8; i++)
        if (keys[i] >= kth) ls += __expf(vals[i] - rowmax);
    sred[tid] = ls;
    __syncthreads();
    for (int s = 128; s > 0; s >>= 1) {
        if (tid < s) sred[tid] += sred[tid + s];
        __syncthreads();
    }
    const float inv = 1.f / sred[0];

    #pragma unroll
    for (int i = 0; i < 8; i++) {
        int idx = tid + (i << 8);
        float p = (keys[i] >= kth) ? __expf(vals[i] - rowmax) * inv : 0.f;
        P[(size_t)row * SEQ + idx] = f2bf(p);
    }
}

// =====================================================================
// outH[m][0:64] = probs[m][:] @ V[:][0:64] (bf16 WMMA, K=2048).
// Double-buffered async staging of probs + V-raw; V transposed in LDS.
// Writes bf16 attn directly (pointer pre-offset by head*64).
// =====================================================================
__global__ __launch_bounds__(128) void pv_kernel(
    const unsigned short* __restrict__ P, const unsigned short* __restrict__ V,
    unsigned short* __restrict__ attn)
{
    __shared__ unsigned ldsA [2][1024];   // probs tile [row][kpair]
    __shared__ unsigned ldsVr[2][1024];   // raw V rows [k][64] bf16
    __shared__ unsigned ldsB [1024];      // [n][kpair] transposed
    const int tid = threadIdx.x;
    const int lane = tid & 31, wv = tid >> 5;
    const int ln = lane & 15, hi = lane >> 4;
    const int m0 = blockIdx.x * 64;
    const unsigned baseA = lds_off_of(&ldsA [0][0]);
    const unsigned baseV = lds_off_of(&ldsVr[0][0]);
    v8f acc[4] = {};

    auto issue = [&](int buf, int k0) {
        #pragma unroll
        for (int rep = 0; rep < 2; rep++) {
            int q = tid + rep * 128;
            int rowA = q >> 2, c4 = q & 3;         // probs: 64 rows x 4 chunks
            async_b128(baseA + (unsigned)(buf * 4096 + q * 16),
                       &P[(size_t)(m0 + rowA) * SEQ + k0 + c4 * 8]);
            int kr = q >> 3, c8 = q & 7;           // V: 32 rows x 8 chunks
            async_b128(baseV + (unsigned)(buf * 4096 + q * 16),
                       &V[(size_t)(k0 + kr) * 64 + c8 * 8]);
        }
    };

    issue(0, 0);
    for (int it = 0; it < (SEQ >> 5); it++) {
        const int buf = it & 1;
        if (it + 1 < (SEQ >> 5)) { issue(buf ^ 1, (it + 1) << 5); wait_async4(); }
        else                     { wait_async0(); }
        __syncthreads();
        const unsigned short* vus = (const unsigned short*)ldsVr[buf]; // [k][64]
        for (int u = tid; u < 1024; u += 128) {
            int n = u >> 4, kp = u & 15;
            unsigned l0 = vus[(2 * kp)     * 64 + n];
            unsigned l1 = vus[(2 * kp + 1) * 64 + n];
            ldsB[u] = l0 | (l1 << 16);
        }
        __syncthreads();
        const unsigned* lA = ldsA[buf];
        FragBF a;
        const int ra = (wv * 16 + ln) * 16;
        #pragma unroll
        for (int i = 0; i < 4; i++) {
            a.u[i]     = lA[ra + hi * 8 + i];
            a.u[4 + i] = lA[ra + hi * 8 + 4 + i];
        }
        #pragma unroll
        for (int j = 0; j < 4; j++) {
            FragBF b;
            const int rb = (j * 16 + ln) * 16 + hi * 8;
            #pragma unroll
            for (int i = 0; i < 8; i++) b.u[i] = ldsB[rb + i];
            acc[j] = __builtin_amdgcn_wmma_f32_16x16x32_bf16(
                false, a.v, false, b.v, (short)0, acc[j], false, false);
        }
        __syncthreads();
    }
    #pragma unroll
    for (int j = 0; j < 4; j++) {
        int col = j * 16 + ln;                       // 0..63 within head
        #pragma unroll
        for (int r = 0; r < 8; r++) {
            int m = m0 + wv * 16 + hi * 8 + r;
            attn[(size_t)m * MODELDIM + col] = f2bf(acc[j][r]);
        }
    }
}

// =====================================================================
extern "C" void kernel_launch(void* const* d_in, const int* in_sizes, int n_in,
                              void* d_out, int out_size, void* d_ws, size_t ws_size,
                              hipStream_t stream) {
    (void)in_sizes; (void)n_in; (void)out_size; (void)ws_size;
    const float* x      = (const float*)d_in[0];
    const float* qkv_w  = (const float*)d_in[1];
    const float* proj_w = (const float*)d_in[2];
    const float* proj_b = (const float*)d_in[3];
    const float* ker_q  = (const float*)d_in[4];
    const float* ker_k  = (const float*)d_in[5];
    const float* ker_v  = (const float*)d_in[6];
    const float* gf     = (const float*)d_in[7];
    const float* tf     = (const float*)d_in[8];
    const float* ph     = (const float*)d_in[9];
    float* out = (float*)d_out;

    // workspace layout (bytes), total ~76 MB
    char* ws = (char*)d_ws;
    float*          qkv    = (float*)(ws);                       // 25,165,824
    unsigned short* qb     = (unsigned short*)(ws + 25165824);   //  4,194,304
    unsigned short* kb     = (unsigned short*)(ws + 29360128);   //  4,194,304
    unsigned short* vb     = (unsigned short*)(ws + 33554432);   //  4,194,304
    float*          scores = (float*)(ws + 37748736);            // 16,777,216
    unsigned short* probs  = (unsigned short*)(ws + 54525952);   //  8,388,608
    unsigned short* xb     = (unsigned short*)(ws + 62914560);   //  4,194,304
    unsigned short* qkvwb  = (unsigned short*)(ws + 67108864);   //  6,291,456
    unsigned short* projwb = (unsigned short*)(ws + 73400320);   //  2,097,152
    unsigned short* attnb  = (unsigned short*)(ws + 75497472);   //  4,194,304

    // 0) one-shot f32 -> bf16 operand conversion
    f32_to_bf16_pk<<<4096, 256, 0, stream>>>(x,      (unsigned*)xb,     SEQ * MODELDIM / 2);
    f32_to_bf16_pk<<<6144, 256, 0, stream>>>(qkv_w,  (unsigned*)qkvwb,  3 * MODELDIM * MODELDIM / 2);
    f32_to_bf16_pk<<<2048, 256, 0, stream>>>(proj_w, (unsigned*)projwb, MODELDIM * MODELDIM / 2);

    // 1) QKV GEMM: (2048x1024) @ (3072x1024)^T
    gemm_bf16_async<<<dim3(48, 32), 128, 0, stream>>>(xb, qkvwb, nullptr, qkv,
                                                      SEQ, 3 * MODELDIM, MODELDIM);
    // 2) dendritic + oscillatory + RoPE -> head-major bf16 Q/K/V
    dendro_rope_kernel<<<4096, 256, 0, stream>>>(qkv, ker_q, ker_k, ker_v,
                                                 gf, tf, ph,
                                                 (unsigned*)qb, (unsigned*)kb,
                                                 (unsigned*)vb);
    // 3) per-head attention (score buffer reused -> sequential heads)
    for (int h = 0; h < NHEADS; h++) {
        const unsigned short* qh = qb + (size_t)h * SEQ * HDIM;
        const unsigned short* kh = kb + (size_t)h * SEQ * HDIM;
        const unsigned short* vh = vb + (size_t)h * SEQ * HDIM;
        scores_kernel<<<dim3(32, 32), 128, 0, stream>>>(qh, kh, scores);
        topk_softmax_kernel<<<SEQ, 256, 0, stream>>>(scores, probs);
        pv_kernel<<<32, 128, 0, stream>>>(probs, vh, attnb + h * HDIM);
    }
    // 4) output projection with bias -> d_out (f32)
    gemm_bf16_async<<<dim3(16, 32), 128, 0, stream>>>(attnb, projwb, proj_b, out,
                                                      SEQ, MODELDIM, MODELDIM);
}
